// MultiHeadAttention_36146444763556
// MI455X (gfx1250) — matrix operand order repaired and checked by reading, not compiled, not run on hardware
//
#include <hip/hip_runtime.h>

typedef __attribute__((ext_vector_type(16))) __bf16          bf16x16;
typedef __attribute__((ext_vector_type(8)))  float           f32x8;
typedef __attribute__((ext_vector_type(8)))  unsigned short  u16x8;
typedef __attribute__((ext_vector_type(4)))  unsigned int    uint4v;
typedef __attribute__((ext_vector_type(8)))  int             int8v;
typedef __attribute__((ext_vector_type(4)))  int             int4v;

#define DEV static __device__ __forceinline__

// ---------- bf16 helpers (round-to-nearest-even) ----------
DEV unsigned short f2bf(float f) {
    unsigned u = __builtin_bit_cast(unsigned, f);
    u += 0x7FFFu + ((u >> 16) & 1u);
    return (unsigned short)(u >> 16);
}

DEV u16x8 ldg8(const unsigned short* p) { return *(const u16x8*)p; }

DEV bf16x16 frag_from(u16x8 lo, u16x8 hi) {
    bf16x16 f;
#pragma unroll
    for (int i = 0; i < 8; ++i) {
        f[i]     = __builtin_bit_cast(__bf16, (unsigned short)lo[i]);
        f[i + 8] = __builtin_bit_cast(__bf16, (unsigned short)hi[i]);
    }
    return f;
}

// A-matrix 16x32 bf16 fragment (ISA 7.12.2): lane m = L&15, half = L>>4,
// element i <-> k = (i>>3)*16 + half*8 + (i&7)  => two contiguous 16B runs.
DEV bf16x16 load_afrag(const unsigned short* base, int ld, int m, int half, int kbase) {
    const unsigned short* r = base + (size_t)m * ld + kbase + half * 8;
    return frag_from(ldg8(r), ldg8(r + 16));
}

// B-matrix 32x16 bf16 fragment: lane n = L&15, half = L>>4,
// element i <-> k = (i>>3)*16 + half*8 + (i&7)  => two contiguous 16B runs from row n of B^T.
DEV bf16x16 load_bfrag(const unsigned short* base, int ld, int n, int half, int kbase) {
    const unsigned short* r = base + (size_t)n * ld + kbase + half * 8;
    return frag_from(ldg8(r), ldg8(r + 16));
}

DEV f32x8 wmma_bf16(bf16x16 a, bf16x16 b, f32x8 c) {
    return __builtin_amdgcn_wmma_f32_16x16x32_bf16(false, a, false, b, (short)0, c, false, false);
}

// ---------------- weight conversion kernels ----------------
// W [H,1024,64] fp32 -> WT [H*64, 1024] bf16 (row = output column, contiguous K)
__global__ void k_cvt_whead(const float* __restrict__ w, unsigned short* __restrict__ wt) {
    const int n = 16 * 64 * 1024;
    for (int o = blockIdx.x * blockDim.x + threadIdx.x; o < n; o += gridDim.x * blockDim.x) {
        int d = o & 1023;
        int e = (o >> 10) & 63;
        int h = o >> 16;
        wt[o] = f2bf(w[(size_t)(h * 1024 + d) * 64 + e]);
    }
}

// Wo [1024,1024] fp32 -> WoT [1024,1024] bf16 transposed
__global__ void k_cvt_wo(const float* __restrict__ w, unsigned short* __restrict__ wt) {
    const int n = 1024 * 1024;
    for (int o = blockIdx.x * blockDim.x + threadIdx.x; o < n; o += gridDim.x * blockDim.x) {
        int d = o & 1023;
        int c = o >> 10;
        wt[o] = f2bf(w[(size_t)d * 1024 + c]);
    }
}

// ---------------- QKV projection ----------------
// Block: one 16-row tile of x, converted f32->bf16 while staging into LDS.
// 8 waves x 24 col-tiles: 192 tiles = {Q,K,V} x 1024 cols, 32 WMMAs each.
// V stored transposed [B*H,64,T] so the PV GEMM's B operand is contiguous.
__global__ void __launch_bounds__(256) k_qkv(
    const float* __restrict__ x,
    const unsigned short* __restrict__ wqT,
    const unsigned short* __restrict__ wkT,
    const unsigned short* __restrict__ wvT,
    const float* __restrict__ bq, const float* __restrict__ bk, const float* __restrict__ bv,
    unsigned short* __restrict__ Qb,   // [B*H, T, 64]
    unsigned short* __restrict__ Kb,   // [B*H, T, 64]
    unsigned short* __restrict__ Vt)   // [B*H, 64, T]
{
    const int T = 2048, D = 1024, H = 16;
    __shared__ unsigned short xt[16 * 1024];

    int blk = blockIdx.x;              // B*T/16 = 512
    int b   = blk / (T / 16);
    int tb  = (blk % (T / 16)) * 16;

    for (int i = threadIdx.x; i < 16 * D; i += 256) {
        int r = i >> 10, c = i & 1023;
        xt[i] = f2bf(x[(size_t)(b * T + tb + r) * D + c]);
    }
    __syncthreads();

    int wave = threadIdx.x >> 5, lane = threadIdx.x & 31;
    int m = lane & 15, half = lane >> 4;

    for (int tile = wave; tile < 192; tile += 8) {
        int mat  = tile / 64;          // 0=Q 1=K 2=V
        int ncol = (tile % 64) * 16;   // output column base (== h*64+e)
        const unsigned short* wt = (mat == 0) ? wqT : (mat == 1) ? wkT : wvT;

        f32x8 acc = {};
        for (int ks = 0; ks < D; ks += 32) {
            bf16x16 a  = load_afrag(xt, D, m, half, ks);
            bf16x16 bb = load_bfrag(wt, D, ncol + m, half, ks);
            acc = wmma_bf16(a, bb, acc);
        }

        const float* bias = (mat == 0) ? bq : (mat == 1) ? bk : bv;
        float bz = bias[ncol + m];
        int h = ncol >> 6, e0 = ncol & 63;

        if (mat < 2) {
            unsigned short* out = (mat == 0) ? Qb : Kb;
            size_t basep = (size_t)(b * H + h) * T * 64;
#pragma unroll
            for (int r = 0; r < 8; ++r) {
                int trow = tb + r + 8 * half;
                out[basep + (size_t)trow * 64 + e0 + m] = f2bf(acc[r] + bz);
            }
        } else {
            size_t rowp = ((size_t)(b * H + h) * 64 + e0 + m) * T + tb + 8 * half;
            u16x8 pk;
#pragma unroll
            for (int r = 0; r < 8; ++r) pk[r] = f2bf(acc[r] + bz);
            *(u16x8*)(Vt + rowp) = pk;
        }
    }
}

// ---------------- flash attention core ----------------
// One wave per 16-query tile; online softmax; C->A layout transpose of P via
// wave-private LDS slice guarded by s_wait_dscnt.
__global__ void __launch_bounds__(256) k_attn(
    const unsigned short* __restrict__ Qb,
    const unsigned short* __restrict__ Kb,
    const unsigned short* __restrict__ Vt,
    const int* __restrict__ maskp,
    unsigned short* __restrict__ attout)   // [B*T, 1024] bf16
{
    const int T = 2048, H = 16;
    __shared__ unsigned short pt[8][512];  // per-wave 16x32 P tile, column-major

    int wave = threadIdx.x >> 5, lane = threadIdx.x & 31;
    int task = blockIdx.x * 8 + wave;      // 8192 tasks
    int qt = task & 127;                   // T/16
    int bh = task >> 7;                    // B*H
    int qb = qt * 16;
    int m = lane & 15, half = lane >> 4;
    int causal = maskp[0];

    const unsigned short* qbase = Qb + (size_t)bh * T * 64 + (size_t)qb * 64;
    const unsigned short* krow  = Kb + (size_t)bh * T * 64;
    const unsigned short* vrow  = Vt + (size_t)bh * 64 * T;

    bf16x16 qa0 = load_afrag(qbase, 64, m, half, 0);
    bf16x16 qa1 = load_afrag(qbase, 64, m, half, 32);

    f32x8 o0 = {}, o1 = {}, o2 = {}, o3 = {};
    float mrow[8], lrow[8];
#pragma unroll
    for (int r = 0; r < 8; ++r) { mrow[r] = -3.0e38f; lrow[r] = 0.0f; }

    int kend = causal ? (qb + 16) : T;
    for (int kb = 0; kb < kend; kb += 32) {
        f32x8 s[2];
#pragma unroll
        for (int t = 0; t < 2; ++t) {
            bf16x16 k0 = load_bfrag(krow, 64, kb + t * 16 + m, half, 0);
            bf16x16 k1 = load_bfrag(krow, 64, kb + t * 16 + m, half, 32);
            f32x8 c = {};
            c = wmma_bf16(qa0, k0, c);
            c = wmma_bf16(qa1, k1, c);
            s[t] = c;
        }
        // scale by sqrt(hd)=8 (faithful to reference) + causal mask (VALU select)
#pragma unroll
        for (int t = 0; t < 2; ++t) {
            int col = kb + t * 16 + m;
#pragma unroll
            for (int r = 0; r < 8; ++r) {
                float v = s[t][r] * 8.0f;
                int rowg = qb + r + 8 * half;
                if (causal && col > rowg) v = -3.0e38f;
                s[t][r] = v;
            }
        }
        // online softmax per row (rows live across 16-lane halves)
        float fac[8];
#pragma unroll
        for (int r = 0; r < 8; ++r) {
            float pm = fmaxf(s[0][r], s[1][r]);
            for (int d = 1; d < 16; d <<= 1) pm = fmaxf(pm, __shfl_xor(pm, d, 32));
            float mn = fmaxf(mrow[r], pm);
            fac[r] = __expf(mrow[r] - mn);
            mrow[r] = mn;
            float p0 = __expf(s[0][r] - mn);
            float p1 = __expf(s[1][r] - mn);
            s[0][r] = p0; s[1][r] = p1;
            float rs = p0 + p1;
            for (int d = 1; d < 16; d <<= 1) rs += __shfl_xor(rs, d, 32);
            lrow[r] = lrow[r] * fac[r] + rs;
        }
#pragma unroll
        for (int r = 0; r < 8; ++r) {
            o0[r] *= fac[r]; o1[r] *= fac[r]; o2[r] *= fac[r]; o3[r] *= fac[r];
        }
        // P (C layout) -> LDS column-major -> A-layout fragment
#pragma unroll
        for (int t = 0; t < 2; ++t) {
            u16x8 pk;
#pragma unroll
            for (int r = 0; r < 8; ++r) pk[r] = f2bf(s[t][r]);
            *(u16x8*)(&pt[wave][(t * 16 + m) * 16 + 8 * half]) = pk;
        }
        asm volatile("s_wait_dscnt 0x0" ::: "memory");
        bf16x16 pa;
#pragma unroll
        for (int i = 0; i < 16; ++i) {
            int k = ((i >> 3) * 16) + half * 8 + (i & 7);
            pa[i] = __builtin_bit_cast(__bf16, pt[wave][k * 16 + m]);
        }
        // O += P @ V  (V^T rows are contiguous)
        bf16x16 v0 = load_bfrag(vrow, T, 0  + m, half, kb);
        bf16x16 v1 = load_bfrag(vrow, T, 16 + m, half, kb);
        bf16x16 v2 = load_bfrag(vrow, T, 32 + m, half, kb);
        bf16x16 v3 = load_bfrag(vrow, T, 48 + m, half, kb);
        o0 = wmma_bf16(pa, v0, o0);
        o1 = wmma_bf16(pa, v1, o1);
        o2 = wmma_bf16(pa, v2, o2);
        o3 = wmma_bf16(pa, v3, o3);
    }

    int b = bh >> 4, h = bh & 15;
#pragma unroll
    for (int r = 0; r < 8; ++r) {
        float inv = 1.0f / lrow[r];
        int rowg = b * T + qb + r + 8 * half;
        size_t base = (size_t)rowg * 1024 + h * 64 + m;
        attout[base +  0] = f2bf(o0[r] * inv);
        attout[base + 16] = f2bf(o1[r] * inv);
        attout[base + 32] = f2bf(o2[r] * inv);
        attout[base + 48] = f2bf(o3[r] * inv);
    }
}

// ---------------- output projection ----------------
// Staging of the (already bf16) activation tile is a pure copy -> use the
// Tensor Data Mover: one 1D tensor_load_to_lds (8192 DWORDs) issued by wave 0,
// completed with s_wait_tensorcnt, then block barrier.
__global__ void __launch_bounds__(256) k_proj(
    const unsigned short* __restrict__ ao,    // [B*T, 1024] bf16
    const unsigned short* __restrict__ woT,   // [1024, 1024] bf16 (transposed)
    const float* __restrict__ bo,
    float* __restrict__ y)                    // [B*T, 1024] f32
{
    const int D = 1024;
    __shared__ unsigned short at[16 * 1024];   // 32 KB tile

    int tb = blockIdx.x * 16;  // flat row base over B*T

    if (threadIdx.x < 32) {    // wave 0 drives the TDM (EXEC ignored by TDM)
        unsigned lds_off = (unsigned)(uintptr_t)(&at[0]);           // LDS aperture: low 32 bits = LDS offset
        unsigned long long ga = (unsigned long long)(uintptr_t)(ao + (size_t)tb * D);
        const unsigned DW = (16u * 1024u * 2u) / 4u;                // 8192 dwords, contiguous

        // D# group 0 (ISA 8.3): count=1 | lds_addr | global_addr | type=2
        uint4v g0;
        g0[0] = 1u;                                                 // count=1, user mode
        g0[1] = lds_off;                                            // lds_addr [63:32]
        g0[2] = (unsigned)(ga & 0xFFFFFFFFu);                       // global_addr [95:64]
        g0[3] = (unsigned)((ga >> 32) & 0x01FFFFFFu) | 0x80000000u; // global_addr hi | type=2<<30

        // D# group 1 (ISA 8.4): data_size=4B, tensor_dim0=tile_dim0=8192, dim1=1
        int8v g1;
        g1[0] = (int)(2u << 16);                                    // data_size=2 (4B)
        g1[1] = (int)((DW & 0xFFFFu) << 16);                        // tensor_dim0[15:0] @ bits 63:48
        g1[2] = (int)(((DW >> 16) & 0xFFFFu) | (1u << 16));         // tensor_dim0 hi | tensor_dim1=1
        g1[3] = (int)((DW & 0xFFFFu) << 16);                        // tile_dim0=8192 @ bits 127:112
        g1[4] = 1;                                                  // tile_dim1=1, tile_dim2=0
        g1[5] = (int)DW;                                            // tensor_dim0_stride lo32
        g1[6] = (int)((DW & 0xFFFFu) << 16);                        // stride hi=0 | dim1_stride lo16
        g1[7] = 0;

        int4v gz4 = {0, 0, 0, 0};                                   // <=2D: groups 2/3 unused
        int8v gz8 = {0, 0, 0, 0, 0, 0, 0, 0};                       // trailing group (unused)
        __builtin_amdgcn_tensor_load_to_lds(g0, g1, gz4, gz4, gz8, 0);
        __builtin_amdgcn_s_wait_tensorcnt(0);
    }
    __syncthreads();

    int wave = threadIdx.x >> 5, lane = threadIdx.x & 31;
    int m = lane & 15, half = lane >> 4;

    for (int nt = wave; nt < 64; nt += 8) {
        int nc = nt * 16;
        f32x8 acc = {};
        for (int ks = 0; ks < D; ks += 32) {
            bf16x16 a  = load_afrag(at, D, m, half, ks);
            bf16x16 bb = load_bfrag(woT, D, nc + m, half, ks);
            acc = wmma_bf16(a, bb, acc);
        }
        float bz = bo[nc + m];
#pragma unroll
        for (int r = 0; r < 8; ++r)
            y[(size_t)(tb + r + 8 * half) * D + nc + m] = acc[r] + bz;
    }
}

// ---------------- launcher ----------------
extern "C" void kernel_launch(void* const* d_in, const int* in_sizes, int n_in,
                              void* d_out, int out_size, void* d_ws, size_t ws_size,
                              hipStream_t stream) {
    const int B = 4, T = 2048, D = 1024, H = 16;
    const float* x  = (const float*)d_in[0];
    const float* Wq = (const float*)d_in[1];
    const float* bq = (const float*)d_in[2];
    const float* Wk = (const float*)d_in[3];
    const float* bk = (const float*)d_in[4];
    const float* Wv = (const float*)d_in[5];
    const float* bv = (const float*)d_in[6];
    const float* Wo = (const float*)d_in[7];
    const float* bo = (const float*)d_in[8];
    const int*   mk = (const int*)d_in[9];
    float* y = (float*)d_out;

    char* w = (char*)d_ws;
    const size_t szAct = (size_t)B * T * D * 2;      // 16.78 MB
    const size_t szW   = (size_t)D * D * 2;          //  2.10 MB
    unsigned short* wqT  = (unsigned short*)w; w += szW;
    unsigned short* wkT  = (unsigned short*)w; w += szW;
    unsigned short* wvT  = (unsigned short*)w; w += szW;
    unsigned short* woT  = (unsigned short*)w; w += szW;
    unsigned short* Qb   = (unsigned short*)w; w += szAct;
    unsigned short* Kb   = (unsigned short*)w; w += szAct;
    unsigned short* Vt   = (unsigned short*)w; w += szAct;
    unsigned short* aout = (unsigned short*)w; w += szAct;

    k_cvt_whead<<<1024, 256, 0, stream>>>(Wq, wqT);
    k_cvt_whead<<<1024, 256, 0, stream>>>(Wk, wkT);
    k_cvt_whead<<<1024, 256, 0, stream>>>(Wv, wvT);
    k_cvt_wo<<<1024, 256, 0, stream>>>(Wo, woT);

    k_qkv<<<B * (T / 16), 256, 0, stream>>>(x, wqT, wkT, wvT, bq, bk, bv, Qb, Kb, Vt);
    k_attn<<<(B * H * (T / 16)) / 8, 256, 0, stream>>>(Qb, Kb, Vt, mk, aout);
    k_proj<<<(B * T) / 16, 256, 0, stream>>>(aout, woT, bo, y);
}
